// PreciseCNN_QNN_Model_22033182228579
// MI455X (gfx1250) — matrix operand order repaired and checked by reading, not compiled, not run on hardware
//
#include <hip/hip_runtime.h>
#include <math.h>

typedef __attribute__((ext_vector_type(16))) _Float16 v16h;
typedef __attribute__((ext_vector_type(8)))  float    v8f;

// ---------------------------------------------------------------------------
// Kernel 0: f32 -> f16 elementwise convert (for GEMM weights).
// ---------------------------------------------------------------------------
__global__ __launch_bounds__(256) void cvt_f16_kernel(
    const float* __restrict__ src, _Float16* __restrict__ dst, int n)
{
    const int i = blockIdx.x * 256 + threadIdx.x;
    if (i < n) dst[i] = (_Float16)src[i];
}

// ---------------------------------------------------------------------------
// Kernel 1: fused conv1+bn1+relu -> conv2+bn2+relu -> conv3+bn3+relu -> 4x4
// avg-pool. One block == one sample; all intermediates live in LDS so the
// 268MB (h1) + 134MB (h2) never touch HBM. Pooled output stored as f16
// (it is only consumed as a WMMA A-operand).
// ---------------------------------------------------------------------------
__global__ __launch_bounds__(256) void conv_stack_kernel(
    const float* __restrict__ x,
    const float* __restrict__ c1w, const float* __restrict__ c1b,
    const float* __restrict__ g1, const float* __restrict__ be1,
    const float* __restrict__ m1, const float* __restrict__ v1,
    const float* __restrict__ c2w, const float* __restrict__ c2b,
    const float* __restrict__ g2, const float* __restrict__ be2,
    const float* __restrict__ m2, const float* __restrict__ v2,
    const float* __restrict__ c3w, const float* __restrict__ c3b,
    const float* __restrict__ g3, const float* __restrict__ be3,
    const float* __restrict__ m3, const float* __restrict__ v3,
    _Float16* __restrict__ pooled)
{
    __shared__ float ldsA[8192];   // h1 [8,32,32] (32KB); later reused for c3w (4608 floats)
    __shared__ float ldsB[4096];   // h2 [16,16,16] (16KB)
    __shared__ float ldsW[1472];   // c1w(200) | s1(8)@200 | sh1(8)@208 | c2w(1152)@216 |
                                   // s2(16)@1368 | sh2(16)@1384 | s3(32)@1400 | sh3(32)@1432
    const int tid = threadIdx.x;
    const int b   = blockIdx.x;

    for (int i = tid; i < 200;  i += 256) ldsW[i]       = c1w[i];
    for (int i = tid; i < 1152; i += 256) ldsW[216 + i] = c2w[i];
    if (tid < 8)  { float s = g1[tid] * rsqrtf(v1[tid] + 1e-5f);
                    ldsW[200 + tid]  = s;
                    ldsW[208 + tid]  = (c1b[tid] - m1[tid]) * s + be1[tid]; }
    if (tid < 16) { float s = g2[tid] * rsqrtf(v2[tid] + 1e-5f);
                    ldsW[1368 + tid] = s;
                    ldsW[1384 + tid] = (c2b[tid] - m2[tid]) * s + be2[tid]; }
    if (tid < 32) { float s = g3[tid] * rsqrtf(v3[tid] + 1e-5f);
                    ldsW[1400 + tid] = s;
                    ldsW[1432 + tid] = (c3b[tid] - m3[tid]) * s + be3[tid]; }
    __syncthreads();

    // ---- conv1: 1->8ch, 5x5, stride2, pad2; 64x64 -> 8x32x32 ----
    const float* xin = x + (size_t)b * 4096;
    for (int o = tid; o < 8192; o += 256) {
        const int c = o >> 10, rem = o & 1023, oy = rem >> 5, ox = rem & 31;
        const float* wk = &ldsW[c * 25];
        const int iy0 = 2 * oy - 2, ix0 = 2 * ox - 2;
        float acc = 0.f;
#pragma unroll
        for (int ky = 0; ky < 5; ++ky) {
            const int iy = iy0 + ky;
            if ((unsigned)iy < 64u) {
#pragma unroll
                for (int kx = 0; kx < 5; ++kx) {
                    const int ix = ix0 + kx;
                    if ((unsigned)ix < 64u)
                        acc = fmaf(xin[iy * 64 + ix], wk[ky * 5 + kx], acc);
                }
            }
        }
        ldsA[o] = fmaxf(fmaf(acc, ldsW[200 + c], ldsW[208 + c]), 0.f);
    }
    __syncthreads();

    // ---- conv2: 8->16ch, 3x3, stride2, pad1; 32x32 -> 16x16x16 ----
    for (int o = tid; o < 4096; o += 256) {
        const int c = o >> 8, rem = o & 255, oy = rem >> 4, ox = rem & 15;
        const int iy0 = 2 * oy - 1, ix0 = 2 * ox - 1;
        float acc = 0.f;
        for (int ic = 0; ic < 8; ++ic) {
            const float* hp = &ldsA[ic * 1024];
            const float* wk = &ldsW[216 + (c * 8 + ic) * 9];
#pragma unroll
            for (int ky = 0; ky < 3; ++ky) {
                const int iy = iy0 + ky;
                if ((unsigned)iy < 32u) {
#pragma unroll
                    for (int kx = 0; kx < 3; ++kx) {
                        const int ix = ix0 + kx;
                        if ((unsigned)ix < 32u)
                            acc = fmaf(hp[iy * 32 + ix], wk[ky * 3 + kx], acc);
                    }
                }
            }
        }
        ldsB[o] = fmaxf(fmaf(acc, ldsW[1368 + c], ldsW[1384 + c]), 0.f);
    }
    __syncthreads();

    // h1 is dead: stage c3w (32x16x3x3 = 4608 floats) into ldsA
    for (int i = tid; i < 4608; i += 256) ldsA[i] = c3w[i];
    __syncthreads();

    // ---- conv3 (16->32ch, 3x3, s1, pad1) + bn + relu + 4x4 avg-pool ----
    for (int pi = tid; pi < 512; pi += 256) {
        const int c = pi >> 4, py = (pi >> 2) & 3, px = pi & 3;
        const float sc = ldsW[1400 + c], sh = ldsW[1432 + c];
        float psum = 0.f;
        for (int yy = 0; yy < 4; ++yy) {
            const int y = py * 4 + yy;
            for (int xx = 0; xx < 4; ++xx) {
                const int x2 = px * 4 + xx;
                float acc = 0.f;
                for (int ic = 0; ic < 16; ++ic) {
                    const float* hp = &ldsB[ic * 256];
                    const float* wk = &ldsA[(c * 16 + ic) * 9];
#pragma unroll
                    for (int ky = 0; ky < 3; ++ky) {
                        const int iy = y - 1 + ky;
                        if ((unsigned)iy < 16u) {
#pragma unroll
                            for (int kx = 0; kx < 3; ++kx) {
                                const int ix = x2 - 1 + kx;
                                if ((unsigned)ix < 16u)
                                    acc = fmaf(hp[iy * 16 + ix], wk[ky * 3 + kx], acc);
                            }
                        }
                    }
                }
                psum += fmaxf(fmaf(acc, sc, sh), 0.f);
            }
        }
        pooled[(size_t)b * 512 + pi] = (_Float16)(psum * (1.f / 16.f));
    }
}

// ---------------------------------------------------------------------------
// Kernel 2: FC layer as WMMA GEMM with f16 operands in memory:
//   out = relu(A[M,K] * W[N,K]^T + bias)
// One wave owns a 16x16 output tile; K stepped by 32 with
// v_wmma_f32_16x16x32_f16. Because operands are f16 and row-major, each
// lane's fragment slice is two contiguous 16B chunks -> b128 loads:
//   A frag: VGPR j holds K=(j>>2)*16+half*8+(j&3)*2  -> f16 offs k0+half*8,
//           k0+16+half*8 (4 dwords each)
//   B frag: VGPR j holds K=(j>>2)*16+half*8+(j&3)*2  -> f16 offs k0+half*8,
//           k0+16+half*8 (4 dwords each)
//   C/D: lane&15 = N; vgpr r -> M = r + 8*(lane>>4)
// ---------------------------------------------------------------------------
template <bool OUT_HALF>
__global__ __launch_bounds__(128) void fc_wmma_kernel(
    const _Float16* __restrict__ A, const _Float16* __restrict__ W,
    const float* __restrict__ bias, void* __restrict__ outv,
    int K, int N)
{
    const int lane = threadIdx.x & 31;
    const int wave = threadIdx.x >> 5;
    const int m0   = blockIdx.x * 16;
    const int n0   = (blockIdx.y * 4 + wave) * 16;
    const int half = lane >> 4;
    const int lr   = lane & 15;

    const _Float16* arow = A + (size_t)(m0 + lr) * K;
    const _Float16* wrow = W + (size_t)(n0 + lr) * K;

    union Frag { v16h h; float4 f[2]; };

    v8f acc = {};
    for (int k0 = 0; k0 < K; k0 += 32) {
        Frag ua, ub;
        ua.f[0] = *reinterpret_cast<const float4*>(arow + k0 + half * 8);
        ua.f[1] = *reinterpret_cast<const float4*>(arow + k0 + 16 + half * 8);
        ub.f[0] = *reinterpret_cast<const float4*>(wrow + k0 + half * 8);
        ub.f[1] = *reinterpret_cast<const float4*>(wrow + k0 + 16 + half * 8);
        if (k0 + 32 < K) {                       // global_prefetch_b8 next K-chunk
            __builtin_prefetch(arow + k0 + 32, 0, 3);
            __builtin_prefetch(wrow + k0 + 32, 0, 3);
        }
        acc = __builtin_amdgcn_wmma_f32_16x16x32_f16(
            false, ua.h, false, ub.h, (short)0, acc, false, false);
    }

    const float bv = bias[n0 + lr];
#pragma unroll
    for (int r = 0; r < 8; ++r) {
        const int m = m0 + r + (half << 3);
        const float v = fmaxf(acc[r] + bv, 0.f);
        if (OUT_HALF)
            ((_Float16*)outv)[(size_t)m * N + n0 + lr] = (_Float16)v;
        else
            ((float*)outv)[(size_t)m * N + n0 + lr] = v;
    }
}

// ---------------------------------------------------------------------------
// Kernel 3: L2-normalize + 8-qubit state-vector sim, one wave per sample.
// State = 256 amplitudes = 8 VGPRs/lane. Index bits [7:3] = lane, [2:0] = reg.
// Wire q <-> bit (7-q). RY butterflies: shfl_xor for lane bits, register ops
// for reg bits. CNOT(q,q+1): conditional swap/permutation. Output <Z_0>.
// ---------------------------------------------------------------------------
__global__ __launch_bounds__(256) void qsim_kernel(
    const float* __restrict__ feats, const float* __restrict__ qw,
    float* __restrict__ qout)
{
    __shared__ float cs[56], sn[56];
    const int tid = threadIdx.x;
    if (tid < 56) { float s, c; sincosf(qw[tid] * 0.5f, &s, &c); cs[tid] = c; sn[tid] = s; }
    __syncthreads();

    const int lane   = tid & 31;
    const int sample = blockIdx.x * 8 + (tid >> 5);
    const float* fp  = feats + (size_t)sample * 256 + lane * 8;
    const float4 f0 = *reinterpret_cast<const float4*>(fp);
    const float4 f1 = *reinterpret_cast<const float4*>(fp + 4);
    float a[8] = {f0.x, f0.y, f0.z, f0.w, f1.x, f1.y, f1.z, f1.w};

    float ss = 0.f;
#pragma unroll
    for (int r = 0; r < 8; ++r) ss += a[r] * a[r];
#pragma unroll
    for (int off = 16; off; off >>= 1) ss += __shfl_xor(ss, off, 32);
    const float inv = 1.f / fmaxf(sqrtf(ss), 1e-12f);
#pragma unroll
    for (int r = 0; r < 8; ++r) a[r] *= inv;

#pragma unroll
    for (int l = 0; l < 7; ++l) {
        // RY(theta_q) on wire q  (bit p = 7-q)
#pragma unroll
        for (int q = 0; q < 8; ++q) {
            const int p = 7 - q;
            const float c = cs[l * 8 + q], s = sn[l * 8 + q];
            if (p >= 3) {                       // lane-level butterfly
                const int lm = 1 << (p - 3);
                const int hi = (lane >> (p - 3)) & 1;
#pragma unroll
                for (int r = 0; r < 8; ++r) {
                    const float pr = __shfl_xor(a[r], lm, 32);
                    a[r] = hi ? fmaf(s, pr, c * a[r]) : fmaf(-s, pr, c * a[r]);
                }
            } else {                            // register-level butterfly
                const int st = 1 << p;
#pragma unroll
                for (int r0 = 0; r0 < 8; ++r0) {
                    if ((r0 & st) == 0) {
                        const float lo = a[r0], hv = a[r0 + st];
                        a[r0]      = c * lo - s * hv;
                        a[r0 + st] = s * lo + c * hv;
                    }
                }
            }
        }
        // CNOT(q, q+1): control bit pc = 7-q, target bit pt = 6-q
#pragma unroll
        for (int q = 0; q < 7; ++q) {
            const int pc = 7 - q, pt = 6 - q;
            if (pc >= 4) {                      // control lane-bit, target lane-bit
                const int ctrl = (lane >> (pc - 3)) & 1;
                const int lm   = 1 << (pt - 3);
#pragma unroll
                for (int r = 0; r < 8; ++r) {
                    const float sw = __shfl_xor(a[r], lm, 32);
                    a[r] = ctrl ? sw : a[r];
                }
            } else if (pc == 3) {               // control lane-bit0, target reg-bit2
                const int ctrl = lane & 1;
#pragma unroll
                for (int r0 = 0; r0 < 4; ++r0) {
                    const float t0 = a[r0], t1 = a[r0 + 4];
                    a[r0]     = ctrl ? t1 : t0;
                    a[r0 + 4] = ctrl ? t0 : t1;
                }
            } else {                            // both register bits (pc in {1,2})
#pragma unroll
                for (int r0 = 0; r0 < 8; ++r0) {
                    if ((r0 & (1 << pc)) && !(r0 & (1 << pt))) {
                        const int r1 = r0 | (1 << pt);
                        const float t0 = a[r0];
                        a[r0] = a[r1];
                        a[r1] = t0;
                    }
                }
            }
        }
    }

    // <Z_0>: index bit7 (= lane bit4) selects sign
    float part = 0.f;
#pragma unroll
    for (int r = 0; r < 8; ++r) part += a[r] * a[r];
    part = (lane & 16) ? -part : part;
#pragma unroll
    for (int off = 16; off; off >>= 1) part += __shfl_xor(part, off, 32);
    if (lane == 0) qout[sample] = part;
}

// ---------------------------------------------------------------------------
// Kernel 4: tiny head MLP (1->64 relu, BN1d, 64->32 relu, 32->16 relu,
// 16->1 sigmoid). One thread per sample; weights staged to LDS.
// ---------------------------------------------------------------------------
__global__ __launch_bounds__(256) void head_kernel(
    const float* __restrict__ qv,
    const float* __restrict__ w1, const float* __restrict__ b1,
    const float* __restrict__ cg, const float* __restrict__ cb,
    const float* __restrict__ cm, const float* __restrict__ cv,
    const float* __restrict__ w2, const float* __restrict__ b2,
    const float* __restrict__ w3, const float* __restrict__ b3,
    const float* __restrict__ w4, const float* __restrict__ b4,
    float* __restrict__ out)
{
    __shared__ float hw1[64], hb1[64], hsc[64], hsh[64];
    __shared__ float hw2[2048], hb2[32], hw3[512], hb3[16], hw4[16], hb4;
    const int tid = threadIdx.x;
    if (tid < 64) {
        hw1[tid] = w1[tid]; hb1[tid] = b1[tid];
        const float s = cg[tid] * rsqrtf(cv[tid] + 1e-5f);
        hsc[tid] = s; hsh[tid] = cb[tid] - cm[tid] * s;
    }
    for (int i = tid; i < 2048; i += 256) hw2[i] = w2[i];
    for (int i = tid; i < 512;  i += 256) hw3[i] = w3[i];
    if (tid < 32) hb2[tid] = b2[tid];
    if (tid < 16) { hb3[tid] = b3[tid]; hw4[tid] = w4[tid]; }
    if (tid == 0) hb4 = b4[0];
    __syncthreads();

    const int i = blockIdx.x * 256 + tid;
    const float q = qv[i];

    float z1[64];
#pragma unroll
    for (int j = 0; j < 64; ++j) {
        const float v = fmaxf(fmaf(q, hw1[j], hb1[j]), 0.f);   // relu first
        z1[j] = fmaf(v, hsc[j], hsh[j]);                        // then BN1d
    }
    float z2[32];
    for (int k = 0; k < 32; ++k) {
        float acc = hb2[k];
#pragma unroll
        for (int j = 0; j < 64; ++j) acc = fmaf(hw2[k * 64 + j], z1[j], acc);
        z2[k] = fmaxf(acc, 0.f);
    }
    float z3[16];
    for (int k = 0; k < 16; ++k) {
        float acc = hb3[k];
#pragma unroll
        for (int j = 0; j < 32; ++j) acc = fmaf(hw3[k * 32 + j], z2[j], acc);
        z3[k] = fmaxf(acc, 0.f);
    }
    float o = hb4;
#pragma unroll
    for (int k = 0; k < 16; ++k) o = fmaf(hw4[k], z3[k], o);
    out[i] = 1.f / (1.f + expf(-o));
}

// ---------------------------------------------------------------------------
extern "C" void kernel_launch(void* const* d_in, const int* in_sizes, int n_in,
                              void* d_out, int out_size, void* d_ws, size_t ws_size,
                              hipStream_t stream)
{
    (void)in_sizes; (void)n_in; (void)out_size; (void)ws_size;
    const float* x    = (const float*)d_in[0];
    const float* c1w  = (const float*)d_in[1];
    const float* c1b  = (const float*)d_in[2];
    const float* bn1g = (const float*)d_in[3];
    const float* bn1b = (const float*)d_in[4];
    const float* bn1m = (const float*)d_in[5];
    const float* bn1v = (const float*)d_in[6];
    const float* c2w  = (const float*)d_in[7];
    const float* c2b  = (const float*)d_in[8];
    const float* bn2g = (const float*)d_in[9];
    const float* bn2b = (const float*)d_in[10];
    const float* bn2m = (const float*)d_in[11];
    const float* bn2v = (const float*)d_in[12];
    const float* c3w  = (const float*)d_in[13];
    const float* c3b  = (const float*)d_in[14];
    const float* bn3g = (const float*)d_in[15];
    const float* bn3b = (const float*)d_in[16];
    const float* bn3m = (const float*)d_in[17];
    const float* bn3v = (const float*)d_in[18];
    const float* f1w  = (const float*)d_in[19];
    const float* f1b  = (const float*)d_in[20];
    const float* f2w  = (const float*)d_in[21];
    const float* f2b  = (const float*)d_in[22];
    const float* qw   = (const float*)d_in[23];
    const float* w1   = (const float*)d_in[24];
    const float* b1   = (const float*)d_in[25];
    const float* cg   = (const float*)d_in[26];
    const float* cb   = (const float*)d_in[27];
    const float* cm   = (const float*)d_in[28];
    const float* cv   = (const float*)d_in[29];
    const float* w2   = (const float*)d_in[30];
    const float* b2   = (const float*)d_in[31];
    const float* w3   = (const float*)d_in[32];
    const float* b3   = (const float*)d_in[33];
    const float* w4   = (const float*)d_in[34];
    const float* b4   = (const float*)d_in[35];

    constexpr int B = 8192;
    // Workspace layout (all region sizes are multiples of 256B):
    _Float16* pooled_h = (_Float16*)d_ws;                    // 8192*512 f16  = 8.4 MB
    _Float16* w1h      = pooled_h + (size_t)B * 512;         // 384*512  f16  = 384 KB
    _Float16* w2h      = w1h + 384 * 512;                    // 256*384  f16  = 192 KB
    _Float16* fc1o_h   = w2h + 256 * 384;                    // 8192*384 f16  = 6.3 MB
    float*    feats    = (float*)(fc1o_h + (size_t)B * 384); // 8192*256 f32  = 8.4 MB
    float*    qout     = feats + (size_t)B * 256;            // 8192 f32

    // Weight conversion (tiny, once per launch)
    cvt_f16_kernel<<<(384 * 512 + 255) / 256, 256, 0, stream>>>(f1w, w1h, 384 * 512);
    cvt_f16_kernel<<<(256 * 384 + 255) / 256, 256, 0, stream>>>(f2w, w2h, 256 * 384);

    conv_stack_kernel<<<B, 256, 0, stream>>>(
        x, c1w, c1b, bn1g, bn1b, bn1m, bn1v,
        c2w, c2b, bn2g, bn2b, bn2m, bn2v,
        c3w, c3b, bn3g, bn3b, bn3m, bn3v, pooled_h);

    // FC1: [8192,512] x [384,512]^T -> f16 [8192,384]; 512 M-tiles, 24 N-tiles
    fc_wmma_kernel<true><<<dim3(B / 16, 6), 128, 0, stream>>>(
        pooled_h, w1h, f1b, fc1o_h, 512, 384);
    // FC2: [8192,384] x [256,384]^T -> f32 [8192,256]; 16 N-tiles
    fc_wmma_kernel<false><<<dim3(B / 16, 4), 128, 0, stream>>>(
        fc1o_h, w2h, f2b, feats, 384, 256);

    qsim_kernel<<<B / 8, 256, 0, stream>>>(feats, qw, qout);

    head_kernel<<<B / 256, 256, 0, stream>>>(
        qout, w1, b1, cg, cb, cm, cv, w2, b2, w3, b3, w4, b4, (float*)d_out);
}
